// QuantConvModule_56684978372708
// MI455X (gfx1250) — 8-bit matrix operand order repaired and checked by reading, compiled, not run on hardware
//
#include <hip/hip_runtime.h>
#include <stdint.h>

typedef int v8i __attribute__((ext_vector_type(8)));

#define HW   12544
#define CIN  64
#define COUT 64

__global__ void __launch_bounds__(256)
quantize_to_nhwc(const float* __restrict__ x, int8_t* __restrict__ q)
{
    __shared__ __align__(16) signed char tile[32 * 64];
    const int b    = blockIdx.x;
    const int n    = b / 392;
    const int pix0 = (b % 392) * 32;
    const int t    = threadIdx.x;

    const float* xb = x + (size_t)n * CIN * HW + pix0;
#pragma unroll
    for (int i = 0; i < 8; ++i) {
        int e = i * 256 + t;
        int c = e >> 5;
        int p = e & 31;
        float v = xb[(size_t)c * HW + p];
        int qv = __float2int_rn(v * 20.0f);
        qv = max(-128, min(127, qv));
        tile[p * 64 + c] = (signed char)qv;
    }
    __syncthreads();
    const uint2* src = (const uint2*)tile;
    uint2* dst = (uint2*)(q + (size_t)(n * HW + pix0) * CIN);
    dst[t] = src[t];
}

__global__ void __launch_bounds__(256)
reorder_weights(const int* __restrict__ w, int8_t* __restrict__ wr)
{
    int idx = blockIdx.x * 256 + threadIdx.x;
    if (idx >= 9 * 4 * 32 * 8) return;
    int v    = idx & 7;
    int lane = (idx >> 3) & 31;
    int ct   = (idx >> 8) & 3;
    int tap  = idx >> 10;
    int kh = tap / 3, kw = tap % 3;
    int cout = ct * 16 + (lane & 15);
    int kb = ((v >> 1) * 16) + ((lane >> 4) * 8) + ((v & 1) * 4);
    unsigned int pack = 0;
#pragma unroll
    for (int byte = 0; byte < 4; ++byte) {
        int cin = kb + byte;
        int wv = w[((cout * CIN + cin) * 3 + kh) * 3 + kw];
        pack |= ((unsigned int)(wv & 0xFF)) << (8 * byte);
    }
    ((unsigned int*)wr)[idx] = pack;
}

__global__ void __launch_bounds__(224)
conv3x3_wmma_iu8(const int8_t* __restrict__ q, const int8_t* __restrict__ wr,
                 const int* __restrict__ bias, const float* __restrict__ deq,
                 float* __restrict__ out)
{
    const int blk  = blockIdx.x;
    const int n    = blk / 112;
    const int oh   = blk % 112;
    const int t    = threadIdx.x;
    const int wave = t >> 5;
    const int lane = t & 31;
    const int ow0  = wave * 16;

    __shared__ __align__(16) float ldsOut[COUT * 112];

    __builtin_prefetch(wr + (size_t)t * 164, 0, 0);

    const int m     = lane & 15;
    const int kbase = (lane >> 4) * 8;

    v8i acc[4] = {};

#pragma unroll
    for (int kh = 0; kh < 3; ++kh) {
        const int ih = oh + kh - 1;
        const bool rowok = (unsigned)ih < 112u;
#pragma unroll
        for (int kw = 0; kw < 3; ++kw) {
            const int iw = ow0 + m + kw - 1;
            const bool ok = rowok && ((unsigned)iw < 112u);
            const int8_t* ap = q + (size_t)(n * HW + ih * 112 + iw) * CIN + kbase;
            int2 a0{0, 0}, a1{0, 0}, a2{0, 0}, a3{0, 0};
            if (ok) {
                a0 = *(const int2*)(ap +  0);
                a1 = *(const int2*)(ap + 16);
                a2 = *(const int2*)(ap + 32);
                a3 = *(const int2*)(ap + 48);
            }
            v8i A = { a0.x, a0.y, a1.x, a1.y, a2.x, a2.y, a3.x, a3.y };

            const int tap = kh * 3 + kw;
#pragma unroll
            for (int ct = 0; ct < 4; ++ct) {
                const int4* bp =
                    (const int4*)(wr + ((size_t)(tap * 4 + ct) * 32 + lane) * 32);
                int4 b0 = bp[0];
                int4 b1 = bp[1];
                v8i B = { b0.x, b0.y, b0.z, b0.w, b1.x, b1.y, b1.z, b1.w };
                acc[ct] = __builtin_amdgcn_wmma_i32_16x16x64_iu8(
                    true, A, true, B, acc[ct], false, false);
            }
        }
    }

#pragma unroll
    for (int ct = 0; ct < 4; ++ct) {
        const int c  = ct * 16 + (lane & 15);
        const int   bc = bias[c];
        const float dc = deq[c];
#pragma unroll
        for (int r = 0; r < 8; ++r) {
            const int mm = r + ((lane >> 4) * 8);
            ldsOut[c * 112 + ow0 + mm] = (float)(acc[ct][r] + bc) * dc;
        }
    }
    __syncthreads();

    float* ob = out + (size_t)n * COUT * HW + oh * 112;
    const float4* ls = (const float4*)ldsOut;
#pragma unroll
    for (int i = 0; i < 8; ++i) {
        int ch  = i * 224 + t;
        int c   = ch / 28;
        int off = ch % 28;
        *(float4*)(ob + (size_t)c * HW + off * 4) = ls[c * 28 + off];
    }
}

extern "C" void kernel_launch(void* const* d_in, const int* in_sizes, int n_in,
                              void* d_out, int out_size, void* d_ws, size_t ws_size,
                              hipStream_t stream)
{
    const float* x   = (const float*)d_in[0];
    const int*   wq  = (const int*)d_in[1];
    const int*   bia = (const int*)d_in[2];
    const float* dq  = (const float*)d_in[3];
    float* out = (float*)d_out;

    const size_t QBYTES = (size_t)32 * HW * CIN;
    int8_t* qbuf = (int8_t*)d_ws;
    int8_t* wbuf = (int8_t*)d_ws + QBYTES;

    quantize_to_nhwc<<<32 * 392, 256, 0, stream>>>(x, qbuf);

    reorder_weights<<<36, 256, 0, stream>>>(wq, wbuf);

    conv3x3_wmma_iu8<<<32 * 112, 224, 0, stream>>>(qbuf, wbuf, bia, dq, out);
}
